// RelationalMemory_79920751444334
// MI455X (gfx1250) — matrix operand order repaired and checked by reading, not compiled, not run on hardware
//
#include <hip/hip_runtime.h>

// ---------------------------------------------------------------------------
// RelationalMemory forward for MI455X (gfx1250), bf16 WMMA pipeline.
// ---------------------------------------------------------------------------

typedef __bf16 bf16;
typedef __attribute__((ext_vector_type(8)))  __bf16 v8bf;
typedef __attribute__((ext_vector_type(16))) __bf16 v16bf;
typedef __attribute__((ext_vector_type(8)))  float  v8f;

#define C_B   256
#define C_T   64
#define C_IN  512
#define C_MS  16
#define C_H   8
#define C_HD  128
#define C_MSZ 1024
#define C_KS  128
#define C_NG  2048

// ---------------------------------------------------------------------------
// Generic batched WMMA GEMM:  C[z] = act(A[z] (MxK) * B[z]^T-stored (NxK) + bias)
//   A: bf16, row m contiguous in K.  addr = A + (z/a_zq)*a_zs1 + (z%a_zq)*a_zs2 + m*lda + k
//   B: bf16, col n contiguous in K.  addr = Bm + (z/b_zq)*b_zs1 + (z%b_zq)*b_zs2 + n*ldb + k
//   Cf (f32): Cf + (z/cf_zq)*cf_zs1 + (z%cf_zq)*cf_zs2 + m*ldcf + n
//   Cb (bf16): Cb + z*cb_zs + (m/cb_qm)*cb_ms1 + (m%cb_qm)*cb_ms2
//                        + (n/cb_qn)*cb_ns1 + (n%cb_qn)*cb_ns2   (supports transposed stores)
// Each wave computes a (16*RM)(M) x 32(N) tile: 2*RM v_wmma_f32_16x16x32_bf16
// per K-step; RM=2 doubles arithmetic intensity (B-frags reused across RM).
// ---------------------------------------------------------------------------
template <int WM, int WN, int RM>
__global__ __launch_bounds__(WM * WN * 32) void wmma_gemm(
    const bf16* __restrict__ A, int a_zq, long a_zs1, long a_zs2, int lda,
    const bf16* __restrict__ Bm, int b_zq, long b_zs1, long b_zs2, int ldb,
    const float* __restrict__ bias,
    float* __restrict__ Cf, int cf_zq, long cf_zs1, long cf_zs2, int ldcf,
    bf16* __restrict__ Cb, long cb_zs,
    int cb_qm, long cb_ms1, long cb_ms2, int cb_qn, long cb_ns1, long cb_ns2,
    int M, int N, int K, int relu) {
  const int tid  = threadIdx.x;
  const int lane = tid & 31;
  const int wave = tid >> 5;
  const int z    = blockIdx.z;
  const int m0   = (blockIdx.y * WM + wave / WN) * (16 * RM);
  const int n0   = (blockIdx.x * WN + wave % WN) * 32;
  if (m0 >= M || n0 >= N) return;  // wave-uniform: EXEC stays all-ones for WMMA

  const int half = lane >> 4;
  const int l15  = lane & 15;

  // A fragment: lane row = m0+l15; elems 0..7 at K=8*half+e, elems 8..15 at K=16+8*half+e
  const bf16* abase = A + (long)(z / a_zq) * a_zs1 + (long)(z % a_zq) * a_zs2
                        + (long)(m0 + l15) * lda + 8 * half;
  // B fragment: lane col = n0(+16)+l15; elems 0..7 at K=8*half+e, elems 8..15 at K=16+8*half+e
  const bf16* bbase = Bm + (long)(z / b_zq) * b_zs1 + (long)(z % b_zq) * b_zs2 + 8 * half;
  const bf16* bcol0 = bbase + (long)(n0 + l15) * ldb;
  const bf16* bcol1 = bbase + (long)(n0 + 16 + l15) * ldb;

  v8f acc0[RM], acc1[RM];
#pragma unroll
  for (int rm = 0; rm < RM; ++rm) {
    acc0[rm] = (v8f){0.f, 0.f, 0.f, 0.f, 0.f, 0.f, 0.f, 0.f};
    acc1[rm] = (v8f){0.f, 0.f, 0.f, 0.f, 0.f, 0.f, 0.f, 0.f};
  }

  for (int k = 0; k < K; k += 32) {
    const v8bf b0l = *(const v8bf*)(bcol0 + k);
    const v8bf b0h = *(const v8bf*)(bcol0 + k + 16);
    const v8bf b1l = *(const v8bf*)(bcol1 + k);
    const v8bf b1h = *(const v8bf*)(bcol1 + k + 16);
    const v16bf bf0 = __builtin_shufflevector(b0l, b0h, 0,1,2,3,4,5,6,7,8,9,10,11,12,13,14,15);
    const v16bf bf1 = __builtin_shufflevector(b1l, b1h, 0,1,2,3,4,5,6,7,8,9,10,11,12,13,14,15);
#pragma unroll
    for (int rm = 0; rm < RM; ++rm) {
      const bf16* arow = abase + (long)(rm * 16) * lda;
      const v8bf alo = *(const v8bf*)(arow + k);
      const v8bf ahi = *(const v8bf*)(arow + k + 16);
      const v16bf af = __builtin_shufflevector(alo, ahi, 0,1,2,3,4,5,6,7,8,9,10,11,12,13,14,15);
      acc0[rm] = __builtin_amdgcn_wmma_f32_16x16x32_bf16(false, af, false, bf0, (short)0, acc0[rm], false, false);
      acc1[rm] = __builtin_amdgcn_wmma_f32_16x16x32_bf16(false, af, false, bf1, (short)0, acc1[rm], false, false);
    }
    __builtin_prefetch(bcol0 + k + 256, 0, 3);  // global_prefetch_b8 of upcoming B
  }

  const int nA = n0 + l15;
  const int nB = n0 + 16 + l15;
  const float bias0 = bias ? bias[nA] : 0.f;
  const float bias1 = bias ? bias[nB] : 0.f;
  const long cfbase = Cf ? ((long)(z / cf_zq) * cf_zs1 + (long)(z % cf_zq) * cf_zs2) : 0;
  const long cbbase = Cb ? (long)z * cb_zs : 0;

#pragma unroll
  for (int rm = 0; rm < RM; ++rm) {
#pragma unroll
    for (int r = 0; r < 8; ++r) {
      const int m = m0 + rm * 16 + r + 8 * half;
      float v0 = acc0[rm][r] + bias0;
      float v1 = acc1[rm][r] + bias1;
      if (relu) { v0 = fmaxf(v0, 0.f); v1 = fmaxf(v1, 0.f); }
      if (Cf) {
        Cf[cfbase + (long)m * ldcf + nA] = v0;
        Cf[cfbase + (long)m * ldcf + nB] = v1;
      }
      if (Cb) {
        const long mo = (long)(m / cb_qm) * cb_ms1 + (long)(m % cb_qm) * cb_ms2;
        Cb[cbbase + mo + (long)(nA / cb_qn) * cb_ns1 + (long)(nA % cb_qn) * cb_ns2] = (bf16)v0;
        Cb[cbbase + mo + (long)(nB / cb_qn) * cb_ns1 + (long)(nB % cb_qn) * cb_ns2] = (bf16)v1;
      }
    }
  }
}

// ---------------------------------------------------------------------------
// f32 [z][K][N] -> bf16 [z][N][K] tiled transpose (weights are read once; this
// pre-pass makes every GEMM B-operand K-contiguous).
// ---------------------------------------------------------------------------
__global__ __launch_bounds__(256) void transpose_to_bf16(
    const float* __restrict__ src, bf16* __restrict__ dst, int K, int N) {
  __shared__ float tile[32][33];
  src += (long)blockIdx.z * K * N;
  dst += (long)blockIdx.z * N * K;
  const int n0 = blockIdx.x * 32, k0 = blockIdx.y * 32;
  const int tx = threadIdx.x & 31, ty = threadIdx.x >> 5;
  for (int i = ty; i < 32; i += 8)
    tile[i][tx] = src[(long)(k0 + i) * N + (n0 + tx)];
  __syncthreads();
  for (int i = ty; i < 32; i += 8)
    dst[(long)(n0 + i) * K + (k0 + tx)] = (bf16)tile[tx][i];
}

__global__ __launch_bounds__(256) void cvt_f32_to_bf16(
    const float* __restrict__ s, bf16* __restrict__ d, long n) {
  long i = (long)blockIdx.x * blockDim.x + threadIdx.x;
  const long stride = (long)gridDim.x * blockDim.x;
  for (; i < n; i += stride) d[i] = (bf16)s[i];
}

__global__ __launch_bounds__(256) void tanh_f32_to_bf16(
    const float* __restrict__ s, bf16* __restrict__ d, long n) {
  long i = (long)blockIdx.x * blockDim.x + threadIdx.x;
  const long stride = (long)gridDim.x * blockDim.x;
  for (; i < n; i += stride) d[i] = (bf16)tanhf(s[i]);
}

// ---------------------------------------------------------------------------
// Softmax over T=64 followed by top-3 mask (mask applied AFTER softmax, no
// renorm — matches reference). One wave (32 lanes) per row, 2 values/lane.
// ---------------------------------------------------------------------------
__global__ __launch_bounds__(256) void softmax_top3(
    const float* __restrict__ S, bf16* __restrict__ O) {
  const int wave = threadIdx.x >> 5, lane = threadIdx.x & 31;
  const long row = (long)blockIdx.x * 8 + wave;
  const float* p = S + row * 64;
  float x0 = p[lane], x1 = p[lane + 32];
  float mx = fmaxf(x0, x1);
  for (int off = 16; off; off >>= 1) mx = fmaxf(mx, __shfl_xor(mx, off));
  const float e0 = __expf(x0 - mx), e1 = __expf(x1 - mx);
  float s = e0 + e1;
  for (int off = 16; off; off >>= 1) s += __shfl_xor(s, off);
  const float inv = 1.f / s;
  const float p0 = e0 * inv, p1 = e1 * inv;

  float a0 = p0, a1 = p1, thr = 0.f;
#pragma unroll
  for (int it = 0; it < 3; ++it) {
    float m = fmaxf(a0, a1);
    for (int off = 16; off; off >>= 1) m = fmaxf(m, __shfl_xor(m, off));
    thr = m;
    unsigned long long bal0 = __ballot(a0 == m);  // wave-uniform result
    if (bal0) {
      if (lane == (int)__builtin_ctzll(bal0)) a0 = -1.f;
    } else {
      unsigned long long bal1 = __ballot(a1 == m);
      if (lane == (int)__builtin_ctzll(bal1)) a1 = -1.f;
    }
  }
  bf16* o = O + row * 64;
  o[lane]      = (p0 >= thr) ? (bf16)p0 : (bf16)0.f;
  o[lane + 32] = (p1 >= thr) ? (bf16)p1 : (bf16)0.f;
}

// ---------------------------------------------------------------------------
// out = LayerNorm(X + R) * gamma + beta ; row length fixed 1024, 256 thr/row.
// ---------------------------------------------------------------------------
__global__ __launch_bounds__(256) void residual_ln(
    const float* __restrict__ X, const float* __restrict__ R,
    const float* __restrict__ gamma, const float* __restrict__ beta,
    float* __restrict__ outF, bf16* __restrict__ outB) {
  const long base = (long)blockIdx.x * 1024;
  const int t = threadIdx.x;
  const float4 xv = *(const float4*)(X + base + t * 4);
  const float4 rv = *(const float4*)(R + base + t * 4);
  const float v0 = xv.x + rv.x, v1 = xv.y + rv.y, v2 = xv.z + rv.z, v3 = xv.w + rv.w;
  float s  = v0 + v1 + v2 + v3;
  float s2 = v0 * v0 + v1 * v1 + v2 * v2 + v3 * v3;
  for (int off = 16; off; off >>= 1) { s += __shfl_xor(s, off); s2 += __shfl_xor(s2, off); }
  __shared__ float sm[8], sq[8], stats[2];
  const int wave = t >> 5, lane = t & 31;
  if (lane == 0) { sm[wave] = s; sq[wave] = s2; }
  __syncthreads();
  if (t == 0) {
    float ts = 0.f, tq = 0.f;
    for (int i = 0; i < 8; ++i) { ts += sm[i]; tq += sq[i]; }
    const float mean = ts * (1.f / 1024.f);
    const float var  = tq * (1.f / 1024.f) - mean * mean;
    stats[0] = mean;
    stats[1] = rsqrtf(var + 1e-5f);
  }
  __syncthreads();
  const float mean = stats[0], invs = stats[1];
  const float4 gv = *(const float4*)(gamma + t * 4);
  const float4 bv = *(const float4*)(beta + t * 4);
  const float y0 = (v0 - mean) * invs * gv.x + bv.x;
  const float y1 = (v1 - mean) * invs * gv.y + bv.y;
  const float y2 = (v2 - mean) * invs * gv.z + bv.z;
  const float y3 = (v3 - mean) * invs * gv.w + bv.w;
  if (outF) *(float4*)(outF + base + t * 4) = make_float4(y0, y1, y2, y3);
  if (outB) {
    bf16* o = outB + base + t * 4;
    o[0] = (bf16)y0; o[1] = (bf16)y1; o[2] = (bf16)y2; o[3] = (bf16)y3;
  }
}

// pooled[b][d] = mean_t relu(rep_w[d] * inp[b][t][d])  -> bf16
__global__ __launch_bounds__(256) void pool_rep(
    const float* __restrict__ inp, const float* __restrict__ rw, bf16* __restrict__ out) {
  const long idx = (long)blockIdx.x * 256 + threadIdx.x;  // B*MSZ
  const int d = (int)(idx & (C_MSZ - 1));
  const float w = rw[d];
  const float* p = inp + (idx >> 10) * ((long)C_T * C_MSZ) + d;
  float acc = 0.f;
#pragma unroll 4
  for (int t = 0; t < C_T; ++t) acc += fmaxf(0.f, w * p[(long)t * C_MSZ]);
  out[idx] = (bf16)(acc * (1.f / (float)C_T));
}

// next_memory = sigmoid(g_in + ib) * tanh(mem2) + sigmoid(g_fg + fb) * memory
__global__ __launch_bounds__(256) void final_gate(
    const float* __restrict__ gates, const float* __restrict__ gin,
    const float* __restrict__ mem2, const float* __restrict__ mem0,
    const float* __restrict__ ib_p, const float* __restrict__ fb_p,
    float* __restrict__ out) {
  const long idx = (long)blockIdx.x * 256 + threadIdx.x;  // B*MS*MSZ
  const int d = (int)(idx & (C_MSZ - 1));
  const long bs = idx >> 10;           // b*MS + s
  const long b  = bs >> 4;
  const long gbase = bs * C_NG;
  const long ginb  = b * C_NG;
  const float gi = gates[gbase + d] + gin[ginb + d] + *ib_p;
  const float gf = gates[gbase + C_MSZ + d] + gin[ginb + C_MSZ + d] + *fb_p;
  const float ig = 1.f / (1.f + __expf(-gi));
  const float fg = 1.f / (1.f + __expf(-gf));
  out[idx] = ig * tanhf(mem2[idx]) + fg * mem0[idx];
}

// ---------------------------------------------------------------------------
extern "C" void kernel_launch(void* const* d_in, const int* in_sizes, int n_in,
                              void* d_out, int out_size, void* d_ws, size_t ws_size,
                              hipStream_t stream) {
  const float* inputs      = (const float*)d_in[0];
  const float* memory      = (const float*)d_in[1];
  const float* Wq          = (const float*)d_in[2];
  const float* bq          = (const float*)d_in[3];
  const float* Wk          = (const float*)d_in[4];
  const float* bk          = (const float*)d_in[5];
  const float* Wv          = (const float*)d_in[6];
  const float* bv          = (const float*)d_in[7];
  const float* Wmlp        = (const float*)d_in[8];
  const float* bmlp        = (const float*)d_in[9];
  const float* g1          = (const float*)d_in[10];
  const float* b1          = (const float*)d_in[11];
  const float* g2          = (const float*)d_in[12];
  const float* b2          = (const float*)d_in[13];
  const float* Win         = (const float*)d_in[14];
  const float* b_in        = (const float*)d_in[15];
  const float* rep_w       = (const float*)d_in[16];
  const float* Wrep        = (const float*)d_in[17];
  const float* brep        = (const float*)d_in[18];
  const float* Wgroup      = (const float*)d_in[19];
  const float* forget_bias = (const float*)d_in[20];
  const float* input_bias  = (const float*)d_in[21];
  float* out = (float*)d_out;
  (void)in_sizes; (void)n_in; (void)out_size; (void)ws_size;

  char* ws = (char*)d_ws;
  size_t off = 0;
  auto alloc = [&](size_t bytes) -> void* {
    void* p = ws + off;
    off += (bytes + 255) & ~(size_t)255;
    return p;
  };

  const long nInp  = (long)C_B * C_T * C_IN;       // 8,388,608
  const long nMem  = (long)C_B * C_MS * C_MSZ;     // 4,194,304
  const long nAct  = (long)C_B * C_T * C_MSZ;      // 16,777,216
  const long nScr  = (long)C_B * C_H * C_MS * C_T; // 2,097,152
  const long nGate = (long)C_B * C_MS * C_NG;      // 8,388,608

  bf16* inputs_bf = (bf16*)alloc(nInp * 2);
  bf16* memory_bf = (bf16*)alloc(nMem * 2);
  bf16* Wint  = (bf16*)alloc((long)C_IN  * C_MSZ * 2);
  bf16* Wqt   = (bf16*)alloc((long)C_MSZ * C_MSZ * 2);
  bf16* Wkt   = (bf16*)alloc((long)C_MSZ * C_MSZ * 2);
  bf16* Wvt   = (bf16*)alloc((long)C_MSZ * C_MSZ * 2);
  bf16* Wmlpt = (bf16*)alloc((long)C_MSZ * C_MSZ * 2);
  bf16* Wrept = (bf16*)alloc((long)C_MSZ * C_NG * 2);
  bf16* Wgt   = (bf16*)alloc((long)C_MS * C_MSZ * C_NG * 2);
  float* inp_f    = (float*)alloc(nAct * 4);
  bf16*  inp_bf   = (bf16*)alloc(nAct * 2);
  bf16*  q_bf     = (bf16*)alloc(nMem * 2);
  bf16*  k_bf     = (bf16*)alloc(nAct * 2);
  bf16*  vT       = (bf16*)alloc(nAct * 2);   // [B][H][HD][T]
  float* scores_f = (float*)alloc(nScr * 4);
  bf16*  scores_bf= (bf16*)alloc(nScr * 2);
  float* att_f    = (float*)alloc(nMem * 4);
  float* mem1_f   = (float*)alloc(nMem * 4);
  bf16*  mem1_bf  = (bf16*)alloc(nMem * 2);
  bf16*  h1_bf    = (bf16*)alloc(nMem * 2);
  float* h2_f     = (float*)alloc(nMem * 4);
  float* mem2_f   = (float*)alloc(nMem * 4);
  bf16*  pooled_bf= (bf16*)alloc((long)C_B * C_MSZ * 2);
  float* gate_in_f= (float*)alloc((long)C_B * C_NG * 4);
  bf16*  tanh_bf  = (bf16*)alloc(nMem * 2);
  float* gates_f  = (float*)alloc(nGate * 4);

  // --- weight transposes (f32 [K,N] -> bf16 [N,K]) and activation converts ---
  transpose_to_bf16<<<dim3(C_MSZ / 32, C_IN / 32, 1), 256, 0, stream>>>(Win, Wint, C_IN, C_MSZ);
  transpose_to_bf16<<<dim3(C_MSZ / 32, C_MSZ / 32, 1), 256, 0, stream>>>(Wq, Wqt, C_MSZ, C_MSZ);
  transpose_to_bf16<<<dim3(C_MSZ / 32, C_MSZ / 32, 1), 256, 0, stream>>>(Wk, Wkt, C_MSZ, C_MSZ);
  transpose_to_bf16<<<dim3(C_MSZ / 32, C_MSZ / 32, 1), 256, 0, stream>>>(Wv, Wvt, C_MSZ, C_MSZ);
  transpose_to_bf16<<<dim3(C_MSZ / 32, C_MSZ / 32, 1), 256, 0, stream>>>(Wmlp, Wmlpt, C_MSZ, C_MSZ);
  transpose_to_bf16<<<dim3(C_NG / 32, C_MSZ / 32, 1), 256, 0, stream>>>(Wrep, Wrept, C_MSZ, C_NG);
  transpose_to_bf16<<<dim3(C_NG / 32, C_MSZ / 32, C_MS), 256, 0, stream>>>(Wgroup, Wgt, C_MSZ, C_NG);
  cvt_f32_to_bf16<<<4096, 256, 0, stream>>>(inputs, inputs_bf, nInp);
  cvt_f32_to_bf16<<<4096, 256, 0, stream>>>(memory, memory_bf, nMem);

  // --- inp = inputs @ Win + b_in  (f32 for pooling, bf16 for k/v GEMMs) ---
  wmma_gemm<4, 2, 2><<<dim3(C_MSZ / 64, (C_B * C_T) / 128, 1), 256, 0, stream>>>(
      inputs_bf, 1, 0, 0, C_IN,
      Wint, 1, 0, 0, C_IN,
      b_in,
      inp_f, 1, 0, 0, C_MSZ,
      inp_bf, 0, 1, C_MSZ, 0, 1, 1, 0,
      C_B * C_T, C_MSZ, C_IN, 0);

  // --- q = memory @ Wq + bq (bf16) ---
  wmma_gemm<4, 2, 2><<<dim3(C_MSZ / 64, (C_B * C_MS) / 128, 1), 256, 0, stream>>>(
      memory_bf, 1, 0, 0, C_MSZ,
      Wqt, 1, 0, 0, C_MSZ,
      bq,
      nullptr, 1, 0, 0, 0,
      q_bf, 0, 1, C_MSZ, 0, 1, 1, 0,
      C_B * C_MS, C_MSZ, C_MSZ, 0);

  // --- k = inp @ Wk + bk (bf16) ---
  wmma_gemm<4, 2, 2><<<dim3(C_MSZ / 64, (C_B * C_T) / 128, 1), 256, 0, stream>>>(
      inp_bf, 1, 0, 0, C_MSZ,
      Wkt, 1, 0, 0, C_MSZ,
      bk,
      nullptr, 1, 0, 0, 0,
      k_bf, 0, 1, C_MSZ, 0, 1, 1, 0,
      C_B * C_T, C_MSZ, C_MSZ, 0);

  // --- v = inp @ Wv + bv, stored transposed as vT[b][h][d][t] (bf16) ---
  wmma_gemm<4, 2, 2><<<dim3(C_MSZ / 64, (C_B * C_T) / 128, 1), 256, 0, stream>>>(
      inp_bf, 1, 0, 0, C_MSZ,
      Wvt, 1, 0, 0, C_MSZ,
      bv,
      nullptr, 1, 0, 0, 0,
      vT, 0,
      /*qm=*/C_T, /*ms1=*/(long)C_H * C_HD * C_T, /*ms2=*/1,
      /*qn=*/C_HD, /*ns1=*/(long)C_HD * C_T, /*ns2=*/C_T,
      C_B * C_T, C_MSZ, C_MSZ, 0);

  // --- scores_raw[z=(b,h)] = q(16x128) @ k^T(128x64), f32 [B,H,MS,T] ---
  wmma_gemm<1, 2, 1><<<dim3(1, 1, C_B * C_H), 64, 0, stream>>>(
      q_bf, C_H, (long)C_MS * C_H * C_KS, C_KS, C_H * C_KS,
      k_bf, C_H, (long)C_T * C_H * C_KS, C_KS, C_H * C_KS,
      nullptr,
      scores_f, 1, (long)C_MS * C_T, 0, C_T,
      nullptr, 0, 1, 0, 0, 1, 0, 0,
      C_MS, C_T, C_KS, 0);

  // --- softmax + top-3 mask -> bf16 ---
  softmax_top3<<<(C_B * C_H * C_MS) / 8, 256, 0, stream>>>(scores_f, scores_bf);

  // --- att[z=(b,h)] = scores(16x64) @ v(64x128), into [B,MS,H*HD] f32 ---
  wmma_gemm<1, 2, 1><<<dim3(C_HD / 64, 1, C_B * C_H), 64, 0, stream>>>(
      scores_bf, 1, (long)C_MS * C_T, 0, C_T,
      vT, 1, (long)C_HD * C_T, 0, C_T,
      nullptr,
      att_f, C_H, (long)C_MS * C_H * C_HD, C_HD, C_H * C_HD,
      nullptr, 0, 1, 0, 0, 1, 0, 0,
      C_MS, C_HD, C_T, 0);

  // --- mem1 = LN(memory + att) ---
  residual_ln<<<C_B * C_MS, 256, 0, stream>>>(memory, att_f, g1, b1, mem1_f, mem1_bf);

  // --- MLP: h1 = relu(mem1 @ Wmlp + bmlp); h2 = relu(h1 @ Wmlp + bmlp) ---
  wmma_gemm<4, 2, 2><<<dim3(C_MSZ / 64, (C_B * C_MS) / 128, 1), 256, 0, stream>>>(
      mem1_bf, 1, 0, 0, C_MSZ,
      Wmlpt, 1, 0, 0, C_MSZ,
      bmlp,
      nullptr, 1, 0, 0, 0,
      h1_bf, 0, 1, C_MSZ, 0, 1, 1, 0,
      C_B * C_MS, C_MSZ, C_MSZ, 1);
  wmma_gemm<4, 2, 2><<<dim3(C_MSZ / 64, (C_B * C_MS) / 128, 1), 256, 0, stream>>>(
      h1_bf, 1, 0, 0, C_MSZ,
      Wmlpt, 1, 0, 0, C_MSZ,
      bmlp,
      h2_f, 1, 0, 0, C_MSZ,
      nullptr, 0, 1, 0, 0, 1, 0, 0,
      C_B * C_MS, C_MSZ, C_MSZ, 1);

  // --- mem2 = LN(mem1 + h2) ---
  residual_ln<<<C_B * C_MS, 256, 0, stream>>>(mem1_f, h2_f, g2, b2, mem2_f, nullptr);

  // --- gate_inputs = mean_t relu(rep_w * inp) @ Wrep + brep ---
  pool_rep<<<(C_B * C_MSZ) / 256, 256, 0, stream>>>(inp_f, rep_w, pooled_bf);
  wmma_gemm<4, 2, 2><<<dim3(C_NG / 64, C_B / 128, 1), 256, 0, stream>>>(
      pooled_bf, 1, 0, 0, C_MSZ,
      Wrept, 1, 0, 0, C_MSZ,
      brep,
      gate_in_f, 1, 0, 0, C_NG,
      nullptr, 0, 1, 0, 0, 1, 0, 0,
      C_B, C_NG, C_MSZ, 0);

  // --- gate_memory[z=s] = tanh(memory)[:,s,:] @ Wgroup[s] -> gates f32 [B,MS,NG] ---
  tanh_f32_to_bf16<<<4096, 256, 0, stream>>>(memory, tanh_bf, nMem);
  wmma_gemm<4, 2, 2><<<dim3(C_NG / 64, C_B / 128, C_MS), 256, 0, stream>>>(
      tanh_bf, 1, C_MSZ, 0, C_MS * C_MSZ,
      Wgt, 1, (long)C_NG * C_MSZ, 0, C_MSZ,
      nullptr,
      gates_f, 1, C_NG, 0, C_MS * C_NG,
      nullptr, 0, 1, 0, 0, 1, 0, 0,
      C_B, C_NG, C_MSZ, 0);

  // --- gating + output ---
  final_gate<<<(C_B * C_MS * C_MSZ) / 256, 256, 0, stream>>>(
      gates_f, gate_in_f, mem2_f, memory, input_bias, forget_bias, out);
}
